// PairwiseScore_72911364817609
// MI455X (gfx1250) — matrix operand order repaired and checked by reading, not compiled, not run on hardware
//
#include <hip/hip_runtime.h>
#include <hip/hip_bf16.h>

typedef __bf16 bf16_t;
typedef bf16_t v16bf __attribute__((ext_vector_type(16)));
typedef float  v8f   __attribute__((ext_vector_type(8)));

#define GDIM   256
#define DDIM   20
#define KPAD   800      // 788 padded to 800 (25 chunks of 32)
#define HPAD   160      // 150 padded to 160 (10 tiles of 16, 5 chunks of 32)
#define KP1    251
#define FPITCH 808      // LDS pitch for features (conflict-free)
#define HPITCH 168      // LDS pitch for hidden (conflict-free)
#define MBLK   64       // pairs per block
#define NTILES 10       // 160/16

// workspace layout (bytes)
#define WS_W1T 0
#define WS_W2T (HPAD * KPAD * 2)             // 256000
#define WS_B1P (WS_W2T + HPAD * HPAD * 2)    // 307200
#define WS_B2P (WS_B1P + HPAD * 4)
#define WS_W3P (WS_B2P + HPAD * 4)

static __device__ __forceinline__ v16bf load16x2(const bf16_t* p0, const bf16_t* p1) {
    union { v16bf v; uint4 q[2]; } u;
    u.q[0] = *reinterpret_cast<const uint4*>(p0);
    u.q[1] = *reinterpret_cast<const uint4*>(p1);
    return u.v;
}

// ---------------- Kernel 0: weight transpose / pad / bf16 convert ----------------
__global__ void prep_kernel(const float* __restrict__ W1, const float* __restrict__ b1,
                            const float* __restrict__ W2, const float* __restrict__ b2,
                            const float* __restrict__ W3,
                            bf16_t* __restrict__ W1t, bf16_t* __restrict__ W2t,
                            float* __restrict__ b1p, float* __restrict__ b2p,
                            float* __restrict__ W3p) {
    int idx = blockIdx.x * 256 + threadIdx.x;
    if (idx < HPAD * KPAD) {
        int n = idx / KPAD, k = idx % KPAD;
        W1t[idx] = (n < 150 && k < 788) ? (bf16_t)W1[(long)k * 150 + n] : (bf16_t)0.0f;
    }
    if (idx < HPAD * HPAD) {
        int n = idx / HPAD, k = idx % HPAD;
        W2t[idx] = (n < 150 && k < 150) ? (bf16_t)W2[(long)k * 150 + n] : (bf16_t)0.0f;
    }
    if (idx < HPAD) {
        b1p[idx] = (idx < 150) ? b1[idx] : 0.0f;
        b2p[idx] = (idx < 150) ? b2[idx] : 0.0f;
        W3p[idx] = (idx < 150) ? W3[idx] : 0.0f;
    }
}

// ---------------- Kernel 1: pairwise feature build + 3-layer MLP via WMMA ----------------
__global__ __launch_bounds__(128) void
pair_mlp_kernel(const float* __restrict__ g_i,
                const float* __restrict__ ms,
                const float* __restrict__ dist_embed,
                const int*   __restrict__ mention_ids,
                const int*   __restrict__ antecedent_ids,
                const int*   __restrict__ distances,
                const bf16_t* __restrict__ W1t,
                const bf16_t* __restrict__ W2t,
                const float* __restrict__ b1p,
                const float* __restrict__ b2p,
                const float* __restrict__ W3p,
                const float* __restrict__ b3,
                float* __restrict__ outPairs,
                int P) {
    __shared__ bf16_t feat[MBLK * FPITCH];   // 103,424 B
    __shared__ bf16_t h_s [MBLK * HPITCH];   //  21,504 B
    __shared__ int s_i[MBLK], s_j[MBLK], s_di[MBLK];

    const int tid  = threadIdx.x;
    const int wave = tid >> 5;
    const int lane = tid & 31;
    const int l15  = lane & 15;
    const int lhi  = lane >> 4;               // 0 or 1
    const long p0  = (long)blockIdx.x * MBLK;

    // ---- per-pair metadata ----
    if (tid < MBLK) {
        long p = p0 + tid; if (p >= P) p = P - 1;
        int d = distances[p];
        s_i[tid]  = mention_ids[p];
        s_j[tid]  = antecedent_ids[p];
        s_di[tid] = (d>=1)+(d>=2)+(d>=3)+(d>=4)+(d>=8)+(d>=16)+(d>=32)+(d>=64);
    }
    __syncthreads();

    // ---- build feature tile [64 x 800] in bf16 ----
    for (int idx = tid; idx < MBLK * GDIM; idx += 128) {
        int m = idx >> 8, k = idx & (GDIM - 1);
        float a = g_i[(long)s_i[m] * GDIM + k];
        float b = g_i[(long)s_j[m] * GDIM + k];
        bf16_t* row = &feat[m * FPITCH];
        row[k]            = (bf16_t)a;
        row[GDIM + k]     = (bf16_t)b;
        row[2 * GDIM + k] = (bf16_t)(a * b);
    }
    for (int idx = tid; idx < MBLK * 32; idx += 128) {
        int m = idx >> 5, k = idx & 31;
        float v = (k < DDIM) ? dist_embed[s_di[m] * DDIM + k] : 0.0f;
        feat[m * FPITCH + 3 * GDIM + k] = (bf16_t)v;
    }
    __syncthreads();

    // ---- GEMM1: [16 x 800] x [800 x 160] ----
    // Per k-step: one deep load clause (A from LDS + 10 B frags from L2-resident
    // weights), single wait, then 10 back-to-back hazard-free WMMAs.
    const bf16_t* abase = &feat[(wave * 16 + l15) * FPITCH + lhi * 8];
    const bf16_t* bkc   = W1t + l15 * KPAD + lhi * 8;

    v8f acc[NTILES] = {};
#pragma unroll 1
    for (int kc = 0; kc < 25; ++kc) {
        v16bf afr = load16x2(abase, abase + 16);
        v16bf bfr[NTILES];
#pragma unroll
        for (int t = 0; t < NTILES; ++t)
            bfr[t] = load16x2(bkc + t * 16 * KPAD, bkc + t * 16 * KPAD + 16);
#pragma unroll
        for (int t = 0; t < NTILES; ++t)
            acc[t] = __builtin_amdgcn_wmma_f32_16x16x32_bf16(
                false, afr, false, bfr[t], (short)0, acc[t], false, false);
        abase += 32;
        bkc   += 32;
    }

    // ---- bias + ReLU -> bf16 h in LDS ----
#pragma unroll
    for (int t = 0; t < NTILES; ++t) {
        float bv = b1p[t * 16 + l15];
#pragma unroll
        for (int r = 0; r < 8; ++r) {
            float x = acc[t][r] + bv;
            x = fmaxf(x, 0.0f);
            h_s[(wave * 16 + lhi * 8 + r) * HPITCH + t * 16 + l15] = (bf16_t)x;
        }
    }
    asm volatile("s_wait_dscnt 0" ::: "memory");

    // ---- GEMM2: [16 x 160] x [160 x 160] ----
    const bf16_t* a2base = &h_s[(wave * 16 + l15) * HPITCH + lhi * 8];
    const bf16_t* b2kc   = W2t + l15 * HPAD + lhi * 8;

    v8f acc2[NTILES] = {};
#pragma unroll 1
    for (int kc = 0; kc < 5; ++kc) {
        v16bf afr = load16x2(a2base, a2base + 16);
        v16bf bfr[NTILES];
#pragma unroll
        for (int t = 0; t < NTILES; ++t)
            bfr[t] = load16x2(b2kc + t * 16 * HPAD, b2kc + t * 16 * HPAD + 16);
#pragma unroll
        for (int t = 0; t < NTILES; ++t)
            acc2[t] = __builtin_amdgcn_wmma_f32_16x16x32_bf16(
                false, afr, false, bfr[t], (short)0, acc2[t], false, false);
        a2base += 32;
        b2kc   += 32;
    }

    // ---- bias + ReLU + dot with W3 (layer 3) in VALU ----
    float part[8];
#pragma unroll
    for (int r = 0; r < 8; ++r) part[r] = 0.0f;
#pragma unroll
    for (int t = 0; t < NTILES; ++t) {
        float bv = b2p[t * 16 + l15];
        float wv = W3p[t * 16 + l15];
#pragma unroll
        for (int r = 0; r < 8; ++r) {
            float x = acc2[t][r] + bv;
            x = fmaxf(x, 0.0f);
            part[r] += x * wv;
        }
    }
    // reduce over the 16 lanes holding the same rows (xor < 16 keeps halves separate)
#pragma unroll
    for (int r = 0; r < 8; ++r) {
#pragma unroll
        for (int off = 8; off > 0; off >>= 1)
            part[r] += __shfl_xor(part[r], off, 32);
    }

    if (l15 == 0) {
        float b3v = b3[0];
#pragma unroll
        for (int r = 0; r < 8; ++r) {
            int  m = wave * 16 + lhi * 8 + r;
            long p = p0 + m;
            if (p < P) {
                float c = part[r] + b3v + ms[s_i[m]] + ms[s_j[m]];
                outPairs[2 * p]     = 0.0f;
                outPairs[2 * p + 1] = c;
            }
        }
    }
}

// ---------------- Kernel 2: per-group softmax into probs ----------------
__global__ void softmax_kernel(const float* __restrict__ pairsOut,
                               float* __restrict__ probs) {
    __shared__ float red[256];
    const int row = blockIdx.x;
    const int c   = threadIdx.x;

    if (row == 0) {
        if (c < KP1) probs[c] = (c == 0) ? 1.0f : 1000.0f;
        return;
    }
    const int L = (row < 250) ? row : 250;
    const long start = (row <= 250) ? (long)(row - 1) * row / 2
                                    : 31125L + (long)(row - 250) * 250;
    float v = -__builtin_inff();
    if (c < L)       v = pairsOut[2 * (start + c) + 1];
    else if (c == L) v = 0.0f;

    red[c] = v;
    __syncthreads();
    for (int s = 128; s > 0; s >>= 1) {
        if (c < s) red[c] = fmaxf(red[c], red[c + s]);
        __syncthreads();
    }
    float mx = red[0];
    __syncthreads();

    float e = (c <= L) ? __expf(v - mx) : 0.0f;
    red[c] = e;
    __syncthreads();
    for (int s = 128; s > 0; s >>= 1) {
        if (c < s) red[c] += red[c + s];
        __syncthreads();
    }
    float sum = red[0];

    if (c < KP1) probs[(long)row * KP1 + c] = (c <= L) ? (e / sum) : 1000.0f;
}

extern "C" void kernel_launch(void* const* d_in, const int* in_sizes, int n_in,
                              void* d_out, int out_size, void* d_ws, size_t ws_size,
                              hipStream_t stream) {
    const float* g_i  = (const float*)d_in[0];
    const float* msc  = (const float*)d_in[1];
    const float* dEmb = (const float*)d_in[2];
    const float* W1   = (const float*)d_in[3];
    const float* b1   = (const float*)d_in[4];
    const float* W2   = (const float*)d_in[5];
    const float* b2   = (const float*)d_in[6];
    const float* W3   = (const float*)d_in[7];
    const float* b3   = (const float*)d_in[8];
    const int*   mid  = (const int*)d_in[9];
    const int*   aid  = (const int*)d_in[10];
    const int*   dst  = (const int*)d_in[11];

    const int P  = in_sizes[9];    // 468625 pairs
    const int Nr = in_sizes[1];    // 2000 rows of probs (= n_groups + 1)

    char* ws = (char*)d_ws;
    bf16_t* W1t = (bf16_t*)(ws + WS_W1T);
    bf16_t* W2t = (bf16_t*)(ws + WS_W2T);
    float*  b1p = (float*)(ws + WS_B1P);
    float*  b2p = (float*)(ws + WS_B2P);
    float*  W3p = (float*)(ws + WS_W3P);

    float* probs    = (float*)d_out;
    float* pairsOut = probs + (long)Nr * KP1;

    prep_kernel<<<(HPAD * KPAD + 255) / 256, 256, 0, stream>>>(
        W1, b1, W2, b2, W3, W1t, W2t, b1p, b2p, W3p);

    pair_mlp_kernel<<<(P + MBLK - 1) / MBLK, 128, 0, stream>>>(
        g_i, msc, dEmb, mid, aid, dst, W1t, W2t, b1p, b2p, W3p, b3, pairsOut, P);

    softmax_kernel<<<Nr, 256, 0, stream>>>(pairsOut, probs);
}
